// BeamDecoder_91293824844546
// MI455X (gfx1250) — matrix operand order repaired and checked by reading, not compiled, not run on hardware
//
#include <hip/hip_runtime.h>
#include <hip/hip_bf16.h>
#include <math.h>

// ---------------- problem constants (from setup_inputs) ----------------
#define BATCH 64
#define BEAMK 4
#define KB    256          // BEAMK*BATCH rows
#define DIM   512
#define HID   512
#define H3    1536
#define VOCAB 50257
#define TMAX  16           // maxtime(15)+1
#define NTILE 256          // logits columns per block
#define NBLK  197          // ceil(VOCAB/NTILE)
#define VPAD  (NBLK*NTILE) // 50432

typedef __attribute__((ext_vector_type(16))) __bf16 v16bf;
typedef __attribute__((ext_vector_type(8)))  __bf16 v8bf;
typedef __attribute__((ext_vector_type(8)))  float  v8f;

static __device__ __forceinline__ float neg_inf() { return -__builtin_inff(); }

// -------- CDNA5 async global->LDS copy (ASYNCcnt-tracked, ISA 15.18.3) --------
static __device__ __forceinline__ unsigned lds_addr_of(const void* p) {
    // low 32 bits of the flat shared-aperture address = LDS byte address
    return (unsigned)(unsigned long long)p;
}
static __device__ __forceinline__ void async_load_b128(unsigned lds_addr, const void* gaddr) {
    asm volatile("global_load_async_to_lds_b128 %0, %1, off"
                 :: "v"(lds_addr), "v"(gaddr)
                 : "memory");
}
static __device__ __forceinline__ void wait_async0() {
    asm volatile("s_wait_asynccnt 0x0" ::: "memory");
}

static __device__ __forceinline__ v16bf make_a_frag(const __bf16* ap) {
    v8bf lo = *(const v8bf*)(ap);
    v8bf hi = *(const v8bf*)(ap + 8);
    return __builtin_shufflevector(lo, hi, 0,1,2,3,4,5,6,7,8,9,10,11,12,13,14,15);
}

// ---------------- 1) W_out fp32 [512,V] -> bf16 transposed [VPAD,512] ----------------
__global__ void k_prep_wt(const float* __restrict__ Wout, __bf16* __restrict__ Wt) {
    long gid = (long)blockIdx.x * blockDim.x + threadIdx.x;
    long total = (long)VPAD * DIM;
    if (gid >= total) return;
    int n = (int)(gid >> 9);
    int k = (int)(gid & 511);
    float v = (n < VOCAB) ? Wout[(long)k * VOCAB + n] : 0.0f;
    Wt[gid] = (__bf16)v;
}

// ---------------- 1b) GRU weights fp32 [512,1536] -> bf16 transposed [1536,512] ----------------
__global__ void k_prep_wg(const float* __restrict__ Wih, const float* __restrict__ Whh,
                          __bf16* __restrict__ Wgi, __bf16* __restrict__ Wgh) {
    long gid = (long)blockIdx.x * blockDim.x + threadIdx.x;
    if (gid >= (long)H3 * DIM) return;
    int n = (int)(gid >> 9);
    int k = (int)(gid & 511);
    Wgi[gid] = (__bf16)Wih[(long)k * H3 + n];
    Wgh[gid] = (__bf16)Whh[(long)k * H3 + n];
}

// ---------------- 2) init state ----------------
__global__ void k_init(const int* __restrict__ xs, float* __restrict__ hcur,
                       int* __restrict__ xt, float* __restrict__ scores,
                       int* __restrict__ seqs) {
    int tid = blockIdx.x * blockDim.x + threadIdx.x;
    if (tid < KB * HID) hcur[tid] = 0.0f;
    if (tid < KB)       xt[tid] = xs[tid & (BATCH - 1)];   // row r = k*B + b
    if (tid < BATCH * BEAMK) scores[tid] = 0.0f;
    if (tid < BATCH * BEAMK * TMAX) seqs[tid] = 0;
}

// ---------------- 3a) gather embeddings + convert h to bf16 ----------------
__global__ void __launch_bounds__(512) k_embed(
        const int* __restrict__ xt, const float* __restrict__ E,
        const float* __restrict__ hcur,
        __bf16* __restrict__ xemb, __bf16* __restrict__ hbf) {
    int r = blockIdx.x, j = threadIdx.x;
    int id = xt[r];
    xemb[(long)r * DIM + j] = (__bf16)E[(long)id * DIM + j];
    hbf[(long)r * DIM + j]  = (__bf16)hcur[(long)r * HID + j];
}

// ---------------- 3b) GRU step via WMMA ----------------
// grid (4, 16): block covers 16 rows x 128 gate-columns; each wave owns a 16x16
// j-tile and accumulates 6 tiles: ir,iz,ig (x_emb x W_ih) and hr,hz,hg (h x W_hh).
__global__ void __launch_bounds__(256) k_gru_wmma(
        const __bf16* __restrict__ xemb, const __bf16* __restrict__ hbf,
        const __bf16* __restrict__ Wgi, const __bf16* __restrict__ Wgh,
        const float* __restrict__ bih, const float* __restrict__ bhh,
        const float* __restrict__ hcur,
        float* __restrict__ hnew, __bf16* __restrict__ abf) {
    __shared__ __bf16 Xs[16 * DIM];   // 16 KB
    __shared__ __bf16 Hs[16 * DIM];   // 16 KB
    int tid = threadIdx.x;
    int m0 = blockIdx.y * 16;

    // async-stage both 16x512 bf16 panels into LDS
    {
        unsigned lx = lds_addr_of(Xs);
        unsigned lh = lds_addr_of(Hs);
        const char* gx = (const char*)(xemb + (long)m0 * DIM);
        const char* gh = (const char*)(hbf + (long)m0 * DIM);
        for (int i = tid; i < (16 * DIM * 2) / 16; i += 256) {  // 1024 x 16B each
            async_load_b128(lx + i * 16, gx + (long)i * 16);
            async_load_b128(lh + i * 16, gh + (long)i * 16);
        }
        wait_async0();
    }
    __syncthreads();

    int wave = tid >> 5, lane = tid & 31;
    int half = lane >> 4, l16 = lane & 15;
    int jb = blockIdx.x * 128 + wave * 16;     // gate-column tile base, j in [0,512)

    v8f acc[6] = {};
    const __bf16* bpi[3];
    const __bf16* bph[3];
    #pragma unroll
    for (int g = 0; g < 3; ++g) {
        bpi[g] = Wgi + (long)(jb + g * HID + l16) * DIM + half * 16;
        bph[g] = Wgh + (long)(jb + g * HID + l16) * DIM + half * 16;
    }
    const __bf16* ax = Xs + l16 * DIM + half * 16;
    const __bf16* ah = Hs + l16 * DIM + half * 16;

    for (int kk = 0; kk < DIM; kk += 32) {
        v16bf a_x = make_a_frag(ax + kk);
        v16bf a_h = make_a_frag(ah + kk);
        #pragma unroll
        for (int g = 0; g < 3; ++g) {
            v16bf bi = *(const v16bf*)(bpi[g] + kk);
            acc[g] = __builtin_amdgcn_wmma_f32_16x16x32_bf16(
                        false, a_x, false, bi, (short)0, acc[g], false, false);
        }
        #pragma unroll
        for (int g = 0; g < 3; ++g) {
            v16bf bh = *(const v16bf*)(bph[g] + kk);
            acc[3 + g] = __builtin_amdgcn_wmma_f32_16x16x32_bf16(
                        false, a_h, false, bh, (short)0, acc[3 + g], false, false);
        }
    }

    // gate nonlinearities, fully in-register per C-layout element
    int j = jb + l16;
    float b_ir = bih[j], b_iz = bih[j + HID], b_ig = bih[j + 2 * HID];
    float b_hr = bhh[j], b_hz = bhh[j + HID], b_hg = bhh[j + 2 * HID];
    #pragma unroll
    for (int i = 0; i < 8; ++i) {
        int m = m0 + i + half * 8;
        float gir = acc[0][i] + b_ir, giz = acc[1][i] + b_iz, gig = acc[2][i] + b_ig;
        float ghr = acc[3][i] + b_hr, ghz = acc[4][i] + b_hz, ghg = acc[5][i] + b_hg;
        float rr = 1.0f / (1.0f + __expf(-(gir + ghr)));
        float zz = 1.0f / (1.0f + __expf(-(giz + ghz)));
        float nn = tanhf(gig + rr * ghg);
        float ho = hcur[(long)m * HID + j];
        float hv = (1.0f - zz) * nn + zz * ho;
        hnew[(long)m * HID + j] = hv;
        abf[(long)m * HID + j] = (__bf16)hv;
    }
}

// ---------------- 4) logits GEMM (WMMA bf16) + per-row partial softmax/top4 ----------------
__global__ void __launch_bounds__(256) k_logits(
        const __bf16* __restrict__ Wt, const __bf16* __restrict__ A,
        const float* __restrict__ bout,
        float* __restrict__ pmax, float* __restrict__ psum,
        float* __restrict__ pval, int* __restrict__ pidx) {
    __shared__ __bf16 As[16 * DIM];        // 16 KB A panel
    __shared__ float  Lg[16 * NTILE];      // 16 KB logits tile
    __shared__ float  rtmp[16][20];
    __shared__ float  cv[16][64];
    __shared__ int    ci[16][64];

    int tid = threadIdx.x;
    int m0 = blockIdx.y * 16;
    int n0 = blockIdx.x * NTILE;

    // async-stage A panel (16 rows x 512 K) into LDS
    {
        unsigned la = lds_addr_of(As);
        const char* ga = (const char*)(A + (long)m0 * DIM);
        for (int i = tid; i < (16 * DIM * 2) / 16; i += 256) {  // 1024 x 16B
            async_load_b128(la + i * 16, ga + (long)i * 16);
        }
        wait_async0();
    }
    __syncthreads();

    int wave = tid >> 5, lane = tid & 31;
    int half = lane >> 4, l16 = lane & 15;

    for (int sub = 0; sub < 2; ++sub) {
        int ntl = wave * 2 + sub;          // n-tile within block (0..15)
        int nb  = n0 + ntl * 16;           // global column base of this tile
        v8f acc = {};
        const __bf16* bp = Wt + (long)(nb + l16) * DIM + half * 16;
        const __bf16* ap = As + l16 * DIM + half * 16;
        for (int kk = 0; kk < DIM; kk += 32) {
            v16bf a = make_a_frag(ap + kk);
            v16bf b = *(const v16bf*)(bp + kk);
            acc = __builtin_amdgcn_wmma_f32_16x16x32_bf16(
                    false, a, false, b, (short)0, acc, false, false);
        }
        // C layout: VGPR i -> M = i + 8*half, N = l16
        #pragma unroll
        for (int i = 0; i < 8; ++i) {
            int m = i + half * 8;
            int n = nb + l16;
            float lv = (n < VOCAB) ? (acc[i] + bout[n]) : neg_inf();
            Lg[m * NTILE + ntl * 16 + l16] = lv;
        }
    }
    __syncthreads();

    // per-row (16 rows) reduce over 256 columns: max, sumexp, top4
    int g = tid >> 4, li = tid & 15;
    int c0 = li * 16;
    float lmax = neg_inf();
    for (int i = 0; i < 16; ++i) lmax = fmaxf(lmax, Lg[g * NTILE + c0 + i]);
    rtmp[g][li] = lmax;
    __syncthreads();
    if (li == 0) {
        float mv = rtmp[g][0];
        for (int i = 1; i < 16; ++i) mv = fmaxf(mv, rtmp[g][i]);
        rtmp[g][16] = mv;
    }
    __syncthreads();
    float rowmax = rtmp[g][16];
    __syncthreads();
    float lsum = 0.0f;
    for (int i = 0; i < 16; ++i) lsum += __expf(Lg[g * NTILE + c0 + i] - rowmax);
    rtmp[g][li] = lsum;
    __syncthreads();
    if (li == 0) {
        float sv = 0.0f;
        for (int i = 0; i < 16; ++i) sv += rtmp[g][i];
        rtmp[g][17] = sv;
    }
    // local top4 over my 16 columns (sorted desc; ties keep earlier/lower index)
    float tv[4] = {neg_inf(), neg_inf(), neg_inf(), neg_inf()};
    int   tix[4] = {0x7FFFFFFF, 0x7FFFFFFF, 0x7FFFFFFF, 0x7FFFFFFF};
    for (int i = 0; i < 16; ++i) {
        float v = Lg[g * NTILE + c0 + i];
        int n = n0 + c0 + i;
        if (v > tv[3]) {
            tv[3] = v; tix[3] = n;
            int p = 3;
            while (p > 0 && tv[p] > tv[p - 1]) {
                float fv = tv[p]; tv[p] = tv[p - 1]; tv[p - 1] = fv;
                int  fi = tix[p]; tix[p] = tix[p - 1]; tix[p - 1] = fi;
                --p;
            }
        }
    }
    #pragma unroll
    for (int s = 0; s < 4; ++s) { cv[g][li * 4 + s] = tv[s]; ci[g][li * 4 + s] = tix[s]; }
    __syncthreads();
    if (li == 0) {
        int r = m0 + g;
        long p = (long)r * NBLK + blockIdx.x;
        pmax[p] = rowmax;
        psum[p] = rtmp[g][17];
        #pragma unroll
        for (int s = 0; s < 4; ++s) {
            int bsel = -1;
            for (int i = 0; i < 64; ++i) {
                if (bsel < 0 || cv[g][i] > cv[g][bsel] ||
                    (cv[g][i] == cv[g][bsel] && ci[g][i] < ci[g][bsel])) bsel = i;
            }
            pval[p * 4 + s] = cv[g][bsel];
            pidx[p * 4 + s] = ci[g][bsel];
            cv[g][bsel] = neg_inf();
            ci[g][bsel] = 0x7FFFFFFF;
        }
    }
}

// ---------------- 5) per-row reduction of NBLK partials ----------------
__global__ void __launch_bounds__(256) k_reduce(
        const float* __restrict__ pmax, const float* __restrict__ psum,
        const float* __restrict__ pval, const int* __restrict__ pidx,
        float* __restrict__ rlse, float* __restrict__ rval, int* __restrict__ ridx) {
    __shared__ float red[256];
    __shared__ float sval[256][4];
    __shared__ int   sidx[256][4];
    int r = blockIdx.x, tid = threadIdx.x;

    float mx = neg_inf(), sm = 0.0f;
    float v4[4] = {neg_inf(), neg_inf(), neg_inf(), neg_inf()};
    int   i4[4] = {0x7FFFFFFF, 0x7FFFFFFF, 0x7FFFFFFF, 0x7FFFFFFF};
    if (tid < NBLK) {
        long p = (long)r * NBLK + tid;
        mx = pmax[p]; sm = psum[p];
        #pragma unroll
        for (int s = 0; s < 4; ++s) { v4[s] = pval[p * 4 + s]; i4[s] = pidx[p * 4 + s]; }
    }
    red[tid] = mx;
    __syncthreads();
    for (int off = 128; off; off >>= 1) {
        if (tid < off) red[tid] = fmaxf(red[tid], red[tid + off]);
        __syncthreads();
    }
    float gmax = red[0];
    __syncthreads();
    red[tid] = (tid < NBLK) ? sm * __expf(mx - gmax) : 0.0f;
    __syncthreads();
    for (int off = 128; off; off >>= 1) {
        if (tid < off) red[tid] += red[tid + off];
        __syncthreads();
    }
    float lse = gmax + __logf(red[0]);

    #pragma unroll
    for (int s = 0; s < 4; ++s) { sval[tid][s] = v4[s]; sidx[tid][s] = i4[s]; }
    for (int off = 128; off; off >>= 1) {
        __syncthreads();
        if (tid < off) {
            float av[8]; int ai[8];
            #pragma unroll
            for (int s = 0; s < 4; ++s) {
                av[s] = sval[tid][s];           ai[s] = sidx[tid][s];
                av[4 + s] = sval[tid + off][s]; ai[4 + s] = sidx[tid + off][s];
            }
            #pragma unroll
            for (int s = 0; s < 4; ++s) {
                int b = 0;
                for (int j = 1; j < 8; ++j) {
                    if (av[j] > av[b] || (av[j] == av[b] && ai[j] < ai[b])) b = j;
                }
                sval[tid][s] = av[b]; sidx[tid][s] = ai[b];
                av[b] = neg_inf(); ai[b] = 0x7FFFFFFF;
            }
        }
    }
    __syncthreads();
    if (tid < 4) {
        rval[(long)r * 4 + tid] = sval[0][tid] - lse;  // logp of top4
        ridx[(long)r * 4 + tid] = sidx[0][tid];
    }
    if (tid == 0) rlse[r] = lse;
}

// ---------------- 6) beam combine + gathers (one block per batch) ----------------
__global__ void __launch_bounds__(256) k_beam(
        const float* __restrict__ rval, const int* __restrict__ ridx,
        float* __restrict__ scores, int* __restrict__ xt, int* __restrict__ seqs,
        const float* __restrict__ hnew, float* __restrict__ hcur, int t) {
    __shared__ float ns[BEAMK];
    __shared__ int   nx[BEAMK];
    __shared__ int   bf[BEAMK];
    __shared__ int   oldseq[BEAMK * TMAX];
    int b = blockIdx.x, tid = threadIdx.x;

    if (tid < BEAMK * TMAX) oldseq[tid] = seqs[b * BEAMK * TMAX + tid];
    if (tid == 0) {
        float tv[4] = {neg_inf(), neg_inf(), neg_inf(), neg_inf()};
        int   tq[4] = {0, 0, 0, 0};
        for (int from = 0; from < BEAMK; ++from) {
            float bs = scores[b * BEAMK + from];
            for (int to = 0; to < BEAMK; ++to) {
                float v = bs + rval[(long)(from * BATCH + b) * BEAMK + to];
                if (v > tv[3]) {                  // strict > keeps earlier flat index on ties
                    tv[3] = v; tq[3] = from * BEAMK + to;
                    int p = 3;
                    while (p > 0 && tv[p] > tv[p - 1]) {
                        float fv = tv[p]; tv[p] = tv[p - 1]; tv[p - 1] = fv;
                        int fq = tq[p]; tq[p] = tq[p - 1]; tq[p - 1] = fq;
                        --p;
                    }
                }
            }
        }
        #pragma unroll
        for (int k = 0; k < BEAMK; ++k) {
            int from = tq[k] / BEAMK, to = tq[k] % BEAMK;
            ns[k] = tv[k];
            bf[k] = from;
            nx[k] = ridx[(long)(from * BATCH + b) * BEAMK + to];
        }
    }
    __syncthreads();
    if (tid < BEAMK * TMAX) {
        int k = tid / TMAX, tt = tid % TMAX;
        int v = oldseq[bf[k] * TMAX + tt];
        if (tt == t) v = nx[k];
        seqs[b * BEAMK * TMAX + tid] = v;
    }
    if (tid < BEAMK) {
        scores[b * BEAMK + tid] = ns[tid];
        xt[tid * BATCH + b] = nx[tid];
    }
    for (int e = tid; e < BEAMK * HID; e += 256) {
        int k = e / HID, d = e % HID;
        hcur[(long)(k * BATCH + b) * HID + d] = hnew[(long)(bf[k] * BATCH + b) * HID + d];
    }
}

// ---------------- 7) emit top beam sequences ----------------
__global__ void k_final(const int* __restrict__ seqs, int* __restrict__ out) {
    int tid = blockIdx.x * blockDim.x + threadIdx.x;
    if (tid < BATCH * TMAX) {
        int b = tid / TMAX, tt = tid % TMAX;
        out[tid] = seqs[b * BEAMK * TMAX + tt];   // beam 0
    }
}

// ---------------- host launcher ----------------
static inline size_t align256(size_t x) { return (x + 255) & ~(size_t)255; }

extern "C" void kernel_launch(void* const* d_in, const int* in_sizes, int n_in,
                              void* d_out, int out_size, void* d_ws, size_t ws_size,
                              hipStream_t stream) {
    const int*   xs   = (const int*)d_in[0];
    const float* E    = (const float*)d_in[1];
    const float* Wih  = (const float*)d_in[2];
    const float* Whh  = (const float*)d_in[3];
    const float* bih  = (const float*)d_in[4];
    const float* bhh  = (const float*)d_in[5];
    const float* Wout = (const float*)d_in[6];
    const float* bout = (const float*)d_in[7];
    (void)in_sizes; (void)n_in; (void)out_size;

    // workspace layout
    size_t off = 0;
    size_t o_wt   = off; off = align256(off + (size_t)VPAD * DIM * 2);
    size_t o_wgi  = off; off = align256(off + (size_t)H3 * DIM * 2);
    size_t o_wgh  = off; off = align256(off + (size_t)H3 * DIM * 2);
    size_t o_hcur = off; off = align256(off + (size_t)KB * HID * 4);
    size_t o_hnew = off; off = align256(off + (size_t)KB * HID * 4);
    size_t o_abf  = off; off = align256(off + (size_t)KB * HID * 2);
    size_t o_xemb = off; off = align256(off + (size_t)KB * DIM * 2);
    size_t o_hbf  = off; off = align256(off + (size_t)KB * HID * 2);
    size_t o_xt   = off; off = align256(off + (size_t)KB * 4);
    size_t o_sc   = off; off = align256(off + (size_t)BATCH * BEAMK * 4);
    size_t o_seq  = off; off = align256(off + (size_t)BATCH * BEAMK * TMAX * 4);
    size_t o_pmax = off; off = align256(off + (size_t)KB * NBLK * 4);
    size_t o_psum = off; off = align256(off + (size_t)KB * NBLK * 4);
    size_t o_pval = off; off = align256(off + (size_t)KB * NBLK * 4 * 4);
    size_t o_pidx = off; off = align256(off + (size_t)KB * NBLK * 4 * 4);
    size_t o_rlse = off; off = align256(off + (size_t)KB * 4);
    size_t o_rval = off; off = align256(off + (size_t)KB * 4 * 4);
    size_t o_ridx = off; off = align256(off + (size_t)KB * 4 * 4);
    if (ws_size < off) return;

    char* ws = (char*)d_ws;
    __bf16* Wt    = (__bf16*)(ws + o_wt);
    __bf16* Wgi   = (__bf16*)(ws + o_wgi);
    __bf16* Wgh   = (__bf16*)(ws + o_wgh);
    float*  hcur  = (float*)(ws + o_hcur);
    float*  hnew  = (float*)(ws + o_hnew);
    __bf16* abf   = (__bf16*)(ws + o_abf);
    __bf16* xemb  = (__bf16*)(ws + o_xemb);
    __bf16* hbf   = (__bf16*)(ws + o_hbf);
    int*    xt    = (int*)(ws + o_xt);
    float*  sc    = (float*)(ws + o_sc);
    int*    seqs  = (int*)(ws + o_seq);
    float*  pmax  = (float*)(ws + o_pmax);
    float*  psum  = (float*)(ws + o_psum);
    float*  pval  = (float*)(ws + o_pval);
    int*    pidx  = (int*)(ws + o_pidx);
    float*  rlse  = (float*)(ws + o_rlse);
    float*  rval  = (float*)(ws + o_rval);
    int*    ridx  = (int*)(ws + o_ridx);

    // one-time (per call) weight converts; all stay L2-resident afterwards
    {
        long total = (long)VPAD * DIM;
        k_prep_wt<<<(int)((total + 255) / 256), 256, 0, stream>>>(Wout, Wt);
        long gtot = (long)H3 * DIM;
        k_prep_wg<<<(int)((gtot + 255) / 256), 256, 0, stream>>>(Wih, Whh, Wgi, Wgh);
    }
    k_init<<<(KB * HID + 255) / 256, 256, 0, stream>>>(xs, hcur, xt, sc, seqs);

    for (int t = 0; t < TMAX; ++t) {
        k_embed<<<KB, 512, 0, stream>>>(xt, E, hcur, xemb, hbf);
        k_gru_wmma<<<dim3(4, 16), 256, 0, stream>>>(xemb, hbf, Wgi, Wgh,
                                                    bih, bhh, hcur, hnew, abf);
        k_logits<<<dim3(NBLK, 16), 256, 0, stream>>>(Wt, abf, bout, pmax, psum, pval, pidx);
        k_reduce<<<KB, 256, 0, stream>>>(pmax, psum, pval, pidx, rlse, rval, ridx);
        k_beam<<<BATCH, 256, 0, stream>>>(rval, ridx, sc, xt, seqs, hnew, hcur, t);
    }
    k_final<<<(BATCH * TMAX + 255) / 256, 256, 0, stream>>>(seqs, (int*)d_out);
}
